// Switch_46170898432170
// MI455X (gfx1250) — 8-bit matrix operand order repaired and checked by reading, compiled, not run on hardware
//
#include <hip/hip_runtime.h>
#include <stdint.h>

typedef __attribute__((ext_vector_type(8))) int   v8i;
typedef __attribute__((ext_vector_type(4))) float v4f;

#define FDIM 512
#define BDIM 128
#define ROWB 520


__global__ void qpack_init_kernel(const float* __restrict__ query,
                                  unsigned char* __restrict__ qp,
                                  unsigned long long* __restrict__ acc)
{
    int i = blockIdx.x * blockDim.x + threadIdx.x;
    if (i < BDIM * FDIM) qp[i] = (query[i] != 0.0f) ? 1u : 0u;
    if (blockIdx.x == 0 && threadIdx.x < BDIM) acc[threadIdx.x] = 0ull;
}

__global__ void __launch_bounds__(256)
match_kernel(const float* __restrict__ keys,
             const float* __restrict__ masks,
             const float* __restrict__ values,
             const float* __restrict__ scores,
             const int*   __restrict__ num_drop_p,
             const unsigned char* __restrict__ qp,
             unsigned long long* __restrict__ acc)
{
    __shared__ unsigned char wtile[16 * ROWB];
    __shared__ int c_lds[16];

    const int t     = threadIdx.x;
    const int kbase = blockIdx.x * 16;

    if (t < 16) c_lds[t] = 0;
    __syncthreads();

    {
        const int row = t >> 4;
        const int f0  = (t & 15) * 32;
        const v4f* krow = (const v4f*)(keys  + (size_t)(kbase + row) * FDIM + f0);
        const v4f* mrow = (const v4f*)(masks + (size_t)(kbase + row) * FDIM + f0);
        int csum = 0;
        #pragma unroll
        for (int j = 0; j < 8; ++j) {
            v4f kf = __builtin_nontemporal_load(krow + j);
            v4f mf = __builtin_nontemporal_load(mrow + j);
            int kb[4] = { kf.x != 0.f, kf.y != 0.f, kf.z != 0.f, kf.w != 0.f };
            int mb[4] = { mf.x != 0.f, mf.y != 0.f, mf.z != 0.f, mf.w != 0.f };
            unsigned w = 0;
            #pragma unroll
            for (int e = 0; e < 4; ++e) {
                int wv = (1 - mb[e]) * (1 - 2 * kb[e]);
                csum  += kb[e] & (1 - mb[e]);
                w     |= ((unsigned)(wv & 0xFF)) << (8 * e);
            }
            *(unsigned*)&wtile[row * ROWB + f0 + 4 * j] = w;
        }
        atomicAdd(&c_lds[row], csum);
    }
    __syncthreads();

    const int lane = t & 31;
    const int wv   = t >> 5;
    const int l15  = lane & 15;
    const int hi   = lane >> 4;

    const unsigned char* abase = qp + (size_t)(wv * 16 + l15) * FDIM;
    const unsigned char* bbase = &wtile[l15 * ROWB];

    const int cc = c_lds[l15];
    v8i acc_v;
    #pragma unroll
    for (int e = 0; e < 8; ++e) acc_v[e] = cc;

    #pragma unroll
    for (int ch = 0; ch < 8; ++ch) {
        union { v8i v; unsigned long long u[4]; } A, Bm;
        const unsigned char* ap = abase + ch * 64 + 8 * hi;
        A.u[0] = *(const unsigned long long*)(ap + 0);
        A.u[1] = *(const unsigned long long*)(ap + 16);
        A.u[2] = *(const unsigned long long*)(ap + 32);
        A.u[3] = *(const unsigned long long*)(ap + 48);
        const unsigned char* bp = bbase + ch * 64 + 8 * hi;
        Bm.u[0] = *(const unsigned long long*)(bp + 0);
        Bm.u[1] = *(const unsigned long long*)(bp + 16);
        Bm.u[2] = *(const unsigned long long*)(bp + 32);
        Bm.u[3] = *(const unsigned long long*)(bp + 48);
        acc_v = __builtin_amdgcn_wmma_i32_16x16x64_iu8(false, A.v, true, Bm.v, acc_v,
                                                       false, false);
    }

    const int nd = *num_drop_p;
    #pragma unroll
    for (int e = 0; e < 8; ++e) {
        if (acc_v[e] == 0) {
            int kg = kbase + l15;
            if (kg >= nd) {
                int rowg = wv * 16 + e + 8 * hi;
                unsigned sb = __float_as_uint(scores[kg]);
                unsigned vb = (values[kg] != 0.0f) ? 1u : 0u;
                unsigned lo = ((0x7FFFFFFFu - (unsigned)(kg - nd)) << 1) | vb;
                unsigned long long packed = ((unsigned long long)sb << 32) | lo;
                atomicMax(&acc[rowg], packed);
            }
        }
    }
}

__global__ void finalize_kernel(const unsigned long long* __restrict__ acc,
                                float* __restrict__ out)
{
    int b = threadIdx.x;
    if (b < BDIM) {
        unsigned long long p = acc[b];
        out[b] = (p != 0ull) ? (float)(p & 1ull) : 0.0f;
    }
}

extern "C" void kernel_launch(void* const* d_in, const int* in_sizes, int n_in,
                              void* d_out, int out_size, void* d_ws, size_t ws_size,
                              hipStream_t stream) {
    const float* query    = (const float*)d_in[0];
    const float* keys     = (const float*)d_in[1];
    const float* masks    = (const float*)d_in[2];
    const float* values   = (const float*)d_in[3];
    const float* scores   = (const float*)d_in[4];
    const int*   num_drop = (const int*)  d_in[6];

    const int K = in_sizes[3];

    unsigned long long* acc = (unsigned long long*)d_ws;
    unsigned char*      qp  = (unsigned char*)d_ws + 1024;

    qpack_init_kernel<<<(BDIM * FDIM + 255) / 256, 256, 0, stream>>>(query, qp, acc);
    match_kernel<<<K / 16, 256, 0, stream>>>(keys, masks, values, scores,
                                             num_drop, qp, acc);
    finalize_kernel<<<1, 128, 0, stream>>>(acc, (float*)d_out);
}
